// VectorQuantizerEMA_80908593922423
// MI455X (gfx1250) — matrix operand order repaired and checked by reading, not compiled, not run on hardware
//
#include <hip/hip_runtime.h>
#include <hip/hip_bf16.h>

// ---- shapes (fixed by the reference) ----
#define B_DIM   32
#define C_DIM   256      // D
#define HW_DIM  1024     // H*W
#define K_CODES 1024
#define N_ROWS  32768    // B*H*W
#define NELEM   8388608  // B*C*H*W

typedef __attribute__((ext_vector_type(8)))  float  v8f;
typedef __attribute__((ext_vector_type(16))) __bf16 v16bf;
typedef __attribute__((ext_vector_type(8)))  __bf16 v8bf;

__device__ __forceinline__ unsigned short f2bf(float x) {
  unsigned u = __float_as_uint(x);
  unsigned r = u + 0x7FFFu + ((u >> 16) & 1u);   // RNE
  return (unsigned short)(r >> 16);
}
__device__ __forceinline__ float bf2f(unsigned short h) {
  return __uint_as_float(((unsigned)h) << 16);
}

// ---------------- kernel 0: codebook prep ----------------
__global__ void __launch_bounds__(256)
vq_prep(const float* __restrict__ embed,
        unsigned short* __restrict__ ehi,
        unsigned short* __restrict__ elo,
        float* __restrict__ e2h,
        int* __restrict__ counts,
        float* __restrict__ lacc) {
  __shared__ float sred[256];
  const int k = blockIdx.x, t = threadIdx.x;
  float x = embed[k * C_DIM + t];
  unsigned short h = f2bf(x);
  ehi[k * C_DIM + t] = h;
  elo[k * C_DIM + t] = f2bf(x - bf2f(h));
  sred[t] = x * x;
  __syncthreads();
  for (int s = 128; s > 0; s >>= 1) {
    if (t < s) sred[t] += sred[t + s];
    __syncthreads();
  }
  if (t == 0) { e2h[k] = 0.5f * sred[0]; counts[k] = 0; }
  if (k == 0 && t == 1) lacc[0] = 0.0f;
}

// ---------------- kernel 1: argmin via bf16 split WMMA ----------------
// Block = 32 query rows (one (b,h), w = 0..31 contiguous), 256 threads = 8 waves.
// Two 16-row A subtiles share every B operand -> 2x B reuse, interleaved
// acc0/acc1 WMMA chains. Each wave scans 8 col-tiles of 16 codes:
// 8 K-chunks x 6 WMMAs per tile.
__global__ void __launch_bounds__(256)
vq_argmin(const float* __restrict__ z_e,
          const unsigned short* __restrict__ ehi,
          const unsigned short* __restrict__ elo,
          const float* __restrict__ e2h,
          int* __restrict__ idxws,
          int* __restrict__ counts,
          float* __restrict__ out_idx) {
  __shared__ unsigned short Ah[32 * C_DIM];   // A tile hi, [m][c], 16 KB
  __shared__ unsigned short Al[32 * C_DIM];   // A tile lo, 16 KB
  __shared__ float rv[8][32];
  __shared__ int   ri[8][32];

  const int t  = threadIdx.x;
  const int n0 = blockIdx.x * 32;

  // ---- stage A tile (strided global -> LDS bf16 hi/lo), 128B coalesced ----
  {
    const int m = t & 31, cs = t >> 5;        // lanes contiguous along w(=m)
    const int b = n0 >> 10, hw = n0 & 1023;
    const float* zb = z_e + ((size_t)b << 18) + hw + m;
#pragma unroll
    for (int it = 0; it < 32; ++it) {
      int c = cs * 32 + it;
      float x = zb[(size_t)c << 10];          // stride C: c*H*W
      unsigned short h = f2bf(x);
      Ah[m * C_DIM + c] = h;
      Al[m * C_DIM + c] = f2bf(x - bf2f(h));
    }
  }
  __syncthreads();

  const int lane = t & 31;
  const int wave = t >> 5;
  const int g  = lane >> 4;   // lane half
  const int nl = lane & 15;   // N index (and A row within subtile)

  float bv0[8], bv1[8]; int bi0[8], bi1[8];
#pragma unroll
  for (int r = 0; r < 8; ++r) {
    bv0[r] = 3.4e38f; bi0[r] = 0;
    bv1[r] = 3.4e38f; bi1[r] = 0;
  }

  const unsigned short* A0H = &Ah[nl * C_DIM];          // subtile 0: rows 0..15
  const unsigned short* A0L = &Al[nl * C_DIM];
  const unsigned short* A1H = &Ah[(16 + nl) * C_DIM];   // subtile 1: rows 16..31
  const unsigned short* A1L = &Al[(16 + nl) * C_DIM];

  for (int kt = wave; kt < K_CODES / 16; kt += 8) {
    const int kcol = kt * 16 + nl;
    const unsigned short* bh = ehi + ((size_t)kcol << 8) + 16 * g;
    const unsigned short* bl = elo + ((size_t)kcol << 8) + 16 * g;
    __builtin_prefetch(bh + 256 * 16 * 8, 0, 1);   // next tile this wave will touch
    v8f acc0 = {}, acc1 = {};
#pragma unroll
    for (int c0 = 0; c0 < C_DIM; c0 += 32) {
      // A operand, 16-bit 16x32 layout: lane-half g holds K in
      // {c0+16g..+15} (contiguous, matches B)  -> two 16B LDS reads each
      v8bf a0h0 = *(const v8bf*)(A0H + c0 + 16 * g);
      v8bf a0h1 = *(const v8bf*)(A0H + c0 + 8 + 16 * g);
      v8bf a0l0 = *(const v8bf*)(A0L + c0 + 16 * g);
      v8bf a0l1 = *(const v8bf*)(A0L + c0 + 8 + 16 * g);
      v8bf a1h0 = *(const v8bf*)(A1H + c0 + 16 * g);
      v8bf a1h1 = *(const v8bf*)(A1H + c0 + 8 + 16 * g);
      v8bf a1l0 = *(const v8bf*)(A1L + c0 + 16 * g);
      v8bf a1l1 = *(const v8bf*)(A1L + c0 + 8 + 16 * g);
      v16bf A0h = __builtin_shufflevector(a0h0, a0h1, 0,1,2,3,4,5,6,7,8,9,10,11,12,13,14,15);
      v16bf A0l = __builtin_shufflevector(a0l0, a0l1, 0,1,2,3,4,5,6,7,8,9,10,11,12,13,14,15);
      v16bf A1h = __builtin_shufflevector(a1h0, a1h1, 0,1,2,3,4,5,6,7,8,9,10,11,12,13,14,15);
      v16bf A1l = __builtin_shufflevector(a1l0, a1l1, 0,1,2,3,4,5,6,7,8,9,10,11,12,13,14,15);
      // B operand, 32x16 layout: lane-half g holds K in {c0+16g..+15}
      // -> one contiguous 32B global read (L2-resident codebook)
      v16bf Bh = *(const v16bf*)(bh + c0);
      v16bf Bl = *(const v16bf*)(bl + c0);
      // split GEMM: x_hi*e_hi + x_hi*e_lo + x_lo*e_hi, fp32 accumulate.
      // acc0/acc1 chains interleave -> no back-to-back RAW on one accumulator.
      acc0 = __builtin_amdgcn_wmma_f32_16x16x32_bf16(false, A0h, false, Bh, (short)0, acc0, false, false);
      acc1 = __builtin_amdgcn_wmma_f32_16x16x32_bf16(false, A1h, false, Bh, (short)0, acc1, false, false);
      acc0 = __builtin_amdgcn_wmma_f32_16x16x32_bf16(false, A0h, false, Bl, (short)0, acc0, false, false);
      acc1 = __builtin_amdgcn_wmma_f32_16x16x32_bf16(false, A1h, false, Bl, (short)0, acc1, false, false);
      acc0 = __builtin_amdgcn_wmma_f32_16x16x32_bf16(false, A0l, false, Bh, (short)0, acc0, false, false);
      acc1 = __builtin_amdgcn_wmma_f32_16x16x32_bf16(false, A1l, false, Bh, (short)0, acc1, false, false);
    }
    // score = 0.5||e||^2 - x.e  (same argmin as full distance)
    float e2v = e2h[kcol];
#pragma unroll
    for (int r = 0; r < 8; ++r) {
      float s0 = e2v - acc0[r];
      float s1 = e2v - acc1[r];
      if (s0 < bv0[r]) { bv0[r] = s0; bi0[r] = kcol; }  // kt ascending -> lowest idx wins ties
      if (s1 < bv1[r]) { bv1[r] = s1; bi1[r] = kcol; }
    }
  }

  // reduce across the 16 lanes of each half (C/D layout: row m = r + 8g)
#pragma unroll
  for (int r = 0; r < 8; ++r) {
    float v0 = bv0[r]; int ix0 = bi0[r];
    float v1 = bv1[r]; int ix1 = bi1[r];
#pragma unroll
    for (int off = 8; off >= 1; off >>= 1) {
      float ov0 = __shfl_xor(v0, off, 32);
      int   oi0 = __shfl_xor(ix0, off, 32);
      float ov1 = __shfl_xor(v1, off, 32);
      int   oi1 = __shfl_xor(ix1, off, 32);
      if (ov0 < v0 || (ov0 == v0 && oi0 < ix0)) { v0 = ov0; ix0 = oi0; }
      if (ov1 < v1 || (ov1 == v1 && oi1 < ix1)) { v1 = ov1; ix1 = oi1; }
    }
    if (nl == 0) {
      rv[wave][r + 8 * g]      = v0; ri[wave][r + 8 * g]      = ix0;
      rv[wave][16 + r + 8 * g] = v1; ri[wave][16 + r + 8 * g] = ix1;
    }
  }
  __syncthreads();

  // reduce across the 8 waves (disjoint code ranges), emit index + histogram
  if (t < 32) {
    float v = rv[0][t]; int ix = ri[0][t];
#pragma unroll
    for (int w = 1; w < 8; ++w) {
      float ov = rv[w][t]; int oi = ri[w][t];
      if (ov < v || (ov == v && oi < ix)) { v = ov; ix = oi; }
    }
    const int n = n0 + t;
    idxws[n]   = ix;
    out_idx[n] = (float)ix;
    atomicAdd(&counts[ix], 1);
  }
}

// ---------------- kernel 2: gather z_q, accumulate MSE ----------------
__global__ void __launch_bounds__(256)
vq_quant(const float* __restrict__ z_e,
         const float* __restrict__ embed,
         const int* __restrict__ idxws,
         float* __restrict__ out_zq,
         float* __restrict__ lacc) {
  __shared__ float sred[256];
  const int t = threadIdx.x;
  const int i = blockIdx.x * 256 + t;
  const int b = i >> 18, c = (i >> 10) & 255, hw = i & 1023;
  const int n = (b << 10) | hw;
  const int k = idxws[n];
  float zq = embed[(size_t)k * C_DIM + c];
  out_zq[i] = zq;                     // z_q_st == z_q numerically
  float d = z_e[i] - zq;
  sred[t] = d * d;
  __syncthreads();
  for (int s = 128; s > 0; s >>= 1) {
    if (t < s) sred[t] += sred[t + s];
    __syncthreads();
  }
  if (t == 0) atomicAdd(lacc, sred[0]);
}

// ---------------- kernel 3: loss + perplexity ----------------
__global__ void __launch_bounds__(1024)
vq_final(const int* __restrict__ counts,
         const float* __restrict__ lacc,
         float* __restrict__ out2) {
  __shared__ float sred[1024];
  const int t = threadIdx.x;
  float p = (float)counts[t] * (1.0f / (float)N_ROWS);
  sred[t] = p * logf(p + 1e-10f);
  __syncthreads();
  for (int s = 512; s > 0; s >>= 1) {
    if (t < s) sred[t] += sred[t + s];
    __syncthreads();
  }
  if (t == 0) {
    out2[0] = 0.25f * lacc[0] * (1.0f / (float)NELEM);  // vq_loss
    out2[1] = expf(-sred[0]);                           // perplexity
  }
}

extern "C" void kernel_launch(void* const* d_in, const int* in_sizes, int n_in,
                              void* d_out, int out_size, void* d_ws, size_t ws_size,
                              hipStream_t stream) {
  const float* z_e   = (const float*)d_in[0];   // (32,256,32,32)
  const float* embed = (const float*)d_in[1];   // (1024,256)
  float* out = (float*)d_out;
  // d_out layout: [0,8388608) z_q_st | [8388608,8421376) indices(float) | loss | perplexity
  char* ws = (char*)d_ws;                       // needs ~1.14 MB
  unsigned short* ehi = (unsigned short*)(ws);              // 512 KB
  unsigned short* elo = (unsigned short*)(ws + 524288);     // 512 KB
  float* e2h   = (float*)(ws + 1048576);                    // 4 KB
  int*   idxws = (int*)  (ws + 1052672);                    // 128 KB
  int*   cnts  = (int*)  (ws + 1183744);                    // 4 KB
  float* lacc  = (float*)(ws + 1187840);                    // 4 B

  vq_prep  <<<K_CODES,      256,  0, stream>>>(embed, ehi, elo, e2h, cnts, lacc);
  vq_argmin<<<N_ROWS / 32,  256,  0, stream>>>(z_e, ehi, elo, e2h, idxws, cnts, out + NELEM);
  vq_quant <<<NELEM / 256,  256,  0, stream>>>(z_e, embed, idxws, out, lacc);
  vq_final <<<1,            1024, 0, stream>>>(cnts, lacc, out + NELEM + N_ROWS);
}
